// FPSSampler_54039278518845
// MI455X (gfx1250) — compile-verified, matrix operand order checked by reading, not run on hardware
//
#include <hip/hip_runtime.h>

#define FPS_N    32768
#define FPS_K    1024
#define THREADS  1024
#define WAVES    32                 // wave32: 1024 threads = 32 waves
#define PPW      (FPS_N / WAVES)    // 1024 points per wave
#define GROUPS   (PPW / 16)         // 64 WMMA groups per wave per round
#define MDREGS   (GROUPS / 2)       // 32 mindist registers per lane
#define CHUNK    8                  // software-pipeline prefetch distance

typedef __attribute__((ext_vector_type(16))) _Float16 v16h;
typedef __attribute__((ext_vector_type(8)))  float    v8f;
typedef __attribute__((ext_vector_type(2)))  float    v2f;
typedef __attribute__((ext_vector_type(8)))  int      v8i;

#if defined(__HIP_DEVICE_COMPILE__) && __has_builtin(__builtin_amdgcn_wmma_f32_16x16x4_f32)
#define HAVE_WMMA_F32X4 1
#else
#define HAVE_WMMA_F32X4 0
#endif

#if defined(__HIP_DEVICE_COMPILE__) && __has_builtin(__builtin_amdgcn_sched_barrier)
#define SCHED_FENCE() __builtin_amdgcn_sched_barrier(0)
#else
#define SCHED_FENCE() do {} while (0)
#endif

// Pin a value in program order: empty asm is side-effecting, so IR passes and
// the scheduler cannot sink the computation that produces it out of its
// scheduling region. Emits no instructions.
#define PIN_F32(v_) asm volatile("" : "+v"(v_))

static __device__ __forceinline__ unsigned pkh2(float a, float b) {
  union { unsigned u; _Float16 h[2]; } t;
  t.h[0] = (_Float16)a;
  t.h[1] = (_Float16)b;
  return t.u;
}

// ---------------------------------------------------------------------------
// Pack kernel: build two f32 planes in workspace.
//   p0[b*N+p] = (x, y)         -> B-fragment rows K=0,1 (lanes 0-15)
//   p1[b*N+p] = (z, |p|^2)     -> B-fragment rows K=2,3 (lanes 16-31)
// ---------------------------------------------------------------------------
__global__ void fps_pack_kernel(const float* __restrict__ x,
                                float2* __restrict__ p0,
                                float2* __restrict__ p1, int total) {
  int i = blockIdx.x * blockDim.x + threadIdx.x;
  if (i >= total) return;
  int b = i >> 15;                 // / FPS_N
  int p = i & (FPS_N - 1);
  const float* xb = x + (size_t)b * (3u * FPS_N);
  float px = xb[p], py = xb[p + FPS_N], pz = xb[p + 2 * FPS_N];
  p0[i] = make_float2(px, py);
  p1[i] = make_float2(pz, px * px + py * py + pz * pz);
}

// ---------------------------------------------------------------------------
// Main FPS kernel, f32 path: V_WMMA_F32_16X16X4_F32.
//   A (16x4, all rows equal) = (-2qx, -2qy, -2qz, 1)
//   B (4x16) column n        = (x_n, y_n, z_n, |p_n|^2)
//   D[m][n] = |p_n|^2 - 2 q.p_n ; d = D + |q|^2
// Points streamed from packed planes (64MB -> L2-resident). Per-group
// prefetch CHUNK ahead; fences every 2 groups + pinned md updates keep
// <=2 D tiles live per region. An opaque per-round *integer* offset defeats
// LICM/CSE while preserving address-space inference (global_load, not flat).
// ---------------------------------------------------------------------------
__global__ __launch_bounds__(THREADS, 1)
void fps_wmma_f32_kernel(const float* __restrict__ x,
                         const float2* __restrict__ p0,
                         const float2* __restrict__ p1,
                         float* __restrict__ out) {
  __shared__ float redv[WAVES];
  __shared__ int   redp[WAVES];
  __shared__ int   bsh;

  const int  b    = blockIdx.x;
  const int  tid  = threadIdx.x;
  const int  lane = tid & 31;
  const int  wave = tid >> 5;
  const bool lo   = (lane < 16);

  const float* xx = x + (size_t)b * (3u * FPS_N);
  const float* xy = xx + FPS_N;
  const float* xz = xx + 2 * FPS_N;
  float*       ob = out + (size_t)b * (3u * FPS_K);

  const int r     = lane & 15;
  const int pbase = wave * PPW + r;
  const int pown  = pbase + (lo ? 0 : MDREGS * 16);  // first owned point

  // lanes 0-15 read (x,y) pairs; lanes 16-31 read (z,sq) pairs
  const float2* plbase = (lo ? p0 : p1) + (size_t)b * FPS_N + pbase;

  float md[MDREGS];
#pragma unroll
  for (int j = 0; j < MDREGS; ++j) md[j] = 1e10f;

  float qx = xx[0], qy = xy[0], qz = xz[0];
  float sqq = qx * qx + qy * qy + qz * qz;
  if (tid == 0) { ob[0] = qx; ob[FPS_K] = qy; ob[2 * FPS_K] = qz; }

  const v8f cz = {};

  for (int t = 1; t < FPS_K; ++t) {
    // Opaque zero offset, redefined every round: blocks LICM/CSE of the
    // point loads while keeping the typed (global) pointer intact.
    unsigned zoff = 0;
    asm volatile("" : "+v"(zoff));
    const float2* plo = plbase + zoff;

    // A fragment (2 VGPRs): lanes 0-15 hold K=0,1; lanes 16-31 hold K=2,3.
    v2f afrag;
    afrag[0] = lo ? (-2.0f * qx) : (-2.0f * qz);
    afrag[1] = lo ? (-2.0f * qy) : 1.0f;

    // preload first CHUNK groups
    float2 u[CHUNK];
#pragma unroll
    for (int i = 0; i < CHUNK; ++i) u[i] = plo[i * 16];

#pragma unroll
    for (int g = 0; g < GROUPS; ++g) {
      const int slot = g % CHUNK;             // compile-time
      float2 cur = u[slot];
      if (g + CHUNK < GROUPS)
        u[slot] = plo[(g + CHUNK) * 16];      // prefetch CHUNK ahead

      v2f bfrag;
      bfrag[0] = cur.x;
      bfrag[1] = cur.y;
#if HAVE_WMMA_F32X4
      v8f c = __builtin_amdgcn_wmma_f32_16x16x4_f32(
          false, afrag, false, bfrag, (short)0, cz, false, false);
#else
      v8f c = cz;                             // only if builtin missing
      c[0] = bfrag[0] * afrag[0];
#endif
      float d = c[0] + sqq;                   // full squared distance

      const bool mine = lo ? (g < MDREGS) : (g >= MDREGS);
      const int  j    = g & (MDREGS - 1);
      float v = fminf(md[j], d);
      md[j] = mine ? v : md[j];
      PIN_F32(md[j]);                         // keep consumer chain in-region

      if ((g & 1) == 1) SCHED_FENCE();        // <=2 live D tiles per region
    }

    // per-lane argmax over owned mindists (first-index tie-break via '>')
    float bestv = -1.0f;
    int   bestoff = 0;
#pragma unroll
    for (int j = 0; j < MDREGS; ++j) {
      bool tk = md[j] > bestv;
      bestv   = tk ? md[j] : bestv;
      bestoff = tk ? (j * 16) : bestoff;      // immediate operand
    }
    int bestp = pown + bestoff;

    // intra-wave reduction
#pragma unroll
    for (int off = 16; off > 0; off >>= 1) {
      float ov = __shfl_xor(bestv, off, 32);
      int   op = __shfl_xor(bestp, off, 32);
      bool take = (ov > bestv) || (ov == bestv && op < bestp);
      bestv = take ? ov : bestv;
      bestp = take ? op : bestp;
    }
    if (lane == 0) { redv[wave] = bestv; redp[wave] = bestp; }
    __syncthreads();

    if (wave == 0) {
      float v = redv[lane];
      int   p = redp[lane];
#pragma unroll
      for (int off = 16; off > 0; off >>= 1) {
        float ov = __shfl_xor(v, off, 32);
        int   op = __shfl_xor(p, off, 32);
        bool take = (ov > v) || (ov == v && op < p);
        v = take ? ov : v;
        p = take ? op : p;
      }
      if (lane == 0) bsh = p;
    }
    __syncthreads();

    const int sel = bsh;
    qx = xx[sel]; qy = xy[sel]; qz = xz[sel];
    sqq = qx * qx + qy * qy + qz * qz;
    if (tid == 0) { ob[t] = qx; ob[FPS_K + t] = qy; ob[2 * FPS_K + t] = qz; }
  }
}

// ---------------------------------------------------------------------------
// Fallback FPS kernel, f16 path: points LDS-resident,
// V_WMMA_F32_16X16X32_F16 with A=(-2q,1) in K=0..3. Same pipeline shape.
// ---------------------------------------------------------------------------
__global__ __launch_bounds__(THREADS, 1)
void fps_wmma_f16_kernel(const float* __restrict__ x, float* __restrict__ out) {
  extern __shared__ unsigned int lds[];
  float* redv = (float*)(lds + 2 * FPS_N);
  int*   redp = (int*)  (lds + 2 * FPS_N + WAVES);
  int*   bsh  = (int*)  (lds + 2 * FPS_N + 2 * WAVES);

  const int  b    = blockIdx.x;
  const int  tid  = threadIdx.x;
  const int  lane = tid & 31;
  const int  wave = tid >> 5;
  const bool lo   = (lane < 16);

  const float* xx = x + (size_t)b * (3u * FPS_N);
  const float* xy = xx + FPS_N;
  const float* xz = xx + 2 * FPS_N;
  float*       ob = out + (size_t)b * (3u * FPS_K);

  uint2* pts = (uint2*)lds;
  for (int p = tid; p < FPS_N; p += THREADS) {
    float px = xx[p], py = xy[p], pz = xz[p];
    float sq = px * px + py * py + pz * pz;
    pts[p] = make_uint2(pkh2(px, py), pkh2(pz, sq));
  }
  __syncthreads();

  float md[MDREGS];
#pragma unroll
  for (int j = 0; j < MDREGS; ++j) md[j] = 1e10f;

  const int r     = lane & 15;
  const int pbase = wave * PPW + r;
  const int pown  = pbase + (lo ? 0 : MDREGS * 16);

  float qx = xx[0], qy = xy[0], qz = xz[0];
  float sqq = qx * qx + qy * qy + qz * qz;
  if (tid == 0) { ob[0] = qx; ob[FPS_K] = qy; ob[2 * FPS_K] = qz; }

  const v8f cz = {};

  for (int t = 1; t < FPS_K; ++t) {
    v8i ai = {};
    int a0 = (int)pkh2(-2.0f * qx, -2.0f * qy);
    int a1 = (int)pkh2(-2.0f * qz, 1.0f);
    ai[0] = lo ? a0 : 0;
    ai[1] = lo ? a1 : 0;
    v16h afrag = __builtin_bit_cast(v16h, ai);

    unsigned zoff = 0;
    asm volatile("" : "+v"(zoff));
    const uint2* plo = pts + pbase + zoff;    // stays addrspace(3) -> ds_load

    uint2 u[CHUNK];
#pragma unroll
    for (int i = 0; i < CHUNK; ++i) u[i] = plo[i * 16];

#pragma unroll
    for (int g = 0; g < GROUPS; ++g) {
      const int slot = g % CHUNK;
      uint2 cur = u[slot];
      if (g + CHUNK < GROUPS)
        u[slot] = plo[(g + CHUNK) * 16];

      v8i bi = {};
      bi[0] = (int)cur.x;
      bi[1] = (int)cur.y;
      v16h bfrag = __builtin_bit_cast(v16h, bi);
      v8f c = __builtin_amdgcn_wmma_f32_16x16x32_f16(
          false, afrag, false, bfrag, (short)0, cz, false, false);
      float d = c[0] + sqq;

      const bool mine = lo ? (g < MDREGS) : (g >= MDREGS);
      const int  j    = g & (MDREGS - 1);
      float v = fminf(md[j], d);
      md[j] = mine ? v : md[j];
      PIN_F32(md[j]);

      if ((g & 1) == 1) SCHED_FENCE();
    }

    float bestv = -1.0f;
    int   bestoff = 0;
#pragma unroll
    for (int j = 0; j < MDREGS; ++j) {
      bool tk = md[j] > bestv;
      bestv   = tk ? md[j] : bestv;
      bestoff = tk ? (j * 16) : bestoff;
    }
    int bestp = pown + bestoff;

#pragma unroll
    for (int off = 16; off > 0; off >>= 1) {
      float ov = __shfl_xor(bestv, off, 32);
      int   op = __shfl_xor(bestp, off, 32);
      bool take = (ov > bestv) || (ov == bestv && op < bestp);
      bestv = take ? ov : bestv;
      bestp = take ? op : bestp;
    }
    if (lane == 0) { redv[wave] = bestv; redp[wave] = bestp; }
    __syncthreads();

    if (wave == 0) {
      float v = redv[lane];
      int   p = redp[lane];
#pragma unroll
      for (int off = 16; off > 0; off >>= 1) {
        float ov = __shfl_xor(v, off, 32);
        int   op = __shfl_xor(p, off, 32);
        bool take = (ov > v) || (ov == v && op < p);
        v = take ? ov : v;
        p = take ? op : p;
      }
      if (lane == 0) bsh[0] = p;
    }
    __syncthreads();

    const int sel = bsh[0];
    qx = xx[sel]; qy = xy[sel]; qz = xz[sel];
    sqq = qx * qx + qy * qy + qz * qz;
    if (tid == 0) { ob[t] = qx; ob[FPS_K + t] = qy; ob[2 * FPS_K + t] = qz; }
  }
}

extern "C" void kernel_launch(void* const* d_in, const int* in_sizes, int n_in,
                              void* d_out, int out_size, void* d_ws, size_t ws_size,
                              hipStream_t stream) {
  const float* x   = (const float*)d_in[0];
  float*       out = (float*)d_out;
  int Btot = in_sizes[0] / (3 * FPS_N);
  if (Btot <= 0) Btot = 1;

  const size_t need = (size_t)Btot * FPS_N * sizeof(float2) * 2;
  if (ws_size >= need) {
    float2* p0 = (float2*)d_ws;
    float2* p1 = p0 + (size_t)Btot * FPS_N;
    int total = Btot * FPS_N;
    fps_pack_kernel<<<dim3((total + 255) / 256), dim3(256), 0, stream>>>(x, p0, p1, total);
    fps_wmma_f32_kernel<<<dim3(Btot), dim3(THREADS), 0, stream>>>(x, p0, p1, out);
  } else {
    const size_t ldsBytes = (size_t)(2 * FPS_N) * sizeof(unsigned int) + 256;
    fps_wmma_f16_kernel<<<dim3(Btot), dim3(THREADS), ldsBytes, stream>>>(x, out);
  }
}
